// DASSM_27771258536650
// MI455X (gfx1250) — matrix operand order repaired and checked by reading, not compiled, not run on hardware
//
#include <hip/hip_runtime.h>
#include <hip/hip_bf16.h>

typedef __attribute__((ext_vector_type(16))) _Float16 v16h;
typedef __attribute__((ext_vector_type(8)))  _Float16 v8h;
typedef __attribute__((ext_vector_type(8)))  float    v8f;

#define N_B   4
#define DM    128
#define HH    128
#define WWD   128
#define HW    (HH*WWD)          // 16384
#define NPIX  (N_B*HW)          // 65536
#define LC    512               // scan chunk length
#define NCH   (HW/LC)           // 32 chunks per batch
#define EPSV  1e-6f
#define RUA   40                // swizzled row unit (halves): 80B = 5*16B aligned,
                                // bank step 20 dwords -> conflict-free over 16 lanes

// A_sw: [ks(4)][row(64)][hi(2)*16 + e]  -> 4*64*40 = 10240 halves (20 KB)
// B_sw: [ks(4)][n(128)][hi(2)*16 + e]   -> 4*128*40 = 20480 halves (40 KB)
#define ASW_HALVES (4 * 64 * RUA)
#define BSW_HALVES (4 * 128 * RUA)

__device__ __forceinline__ void a_swizzle(int k32, int& hi, int& e)
{
    if (k32 < 16) { hi = k32 >> 3; e = k32 & 7; }
    else { int t = k32 - 16; hi = t >> 3; e = 8 + (t & 7); }
}

__device__ __forceinline__ v16h frag_load(const _Float16* p)
{
    v8h a0 = *(const v8h*)(p);       // 16B-aligned -> ds_load_b128
    v8h a1 = *(const v8h*)(p + 8);   // ds_load_b128
    return __builtin_shufflevector(a0, a1, 0,1,2,3,4,5,6,7,8,9,10,11,12,13,14,15);
}

// =====================================================================
// K1: in_proj 1x1 conv as WMMA GEMM.  x: NCHW f32 -> xin: channels-last f32
// =====================================================================
__global__ __launch_bounds__(128) void k_inproj(const float* __restrict__ x,
                                                const float* __restrict__ w,
                                                float* __restrict__ out)
{
    __shared__ _Float16 A_sw[ASW_HALVES];
    __shared__ _Float16 B_sw[BSW_HALVES];
    const int tid = threadIdx.x;
    const int p0  = blockIdx.x * 64;
    const int b   = p0 >> 14;
    const int hw0 = p0 & (HW - 1);

    // stage A pre-swizzled: element (m=i, k=c) -> A_sw[ks][i][hi*16+e]
    for (int idx = tid; idx < 64 * 128; idx += 128) {
        int c = idx >> 6, i = idx & 63;
        float v = x[(b * DM + c) * HW + hw0 + i];           // coalesced along i
        int hi, e; a_swizzle(c & 31, hi, e);
        A_sw[((c >> 5) * 64 + i) * RUA + hi * 16 + e] = (_Float16)v;
    }
    // stage B pre-swizzled: element (k, n) = w[n*128+k] -> B_sw[ks][n][hi*16+e]
    for (int idx = tid; idx < 128 * 128; idx += 128) {
        int n = idx >> 7, k = idx & 127;
        int k32 = k & 31, hi, e; a_swizzle(k32, hi, e);
        B_sw[((k >> 5) * 128 + n) * RUA + hi * 16 + e] = (_Float16)w[idx];
    }
    __syncthreads();

    const int lane = tid & 31, wv = tid >> 5;
    const int m0 = wv * 16;
    const int hi = lane >> 4, lo = lane & 15;

    const v8f vzero = {0.f, 0.f, 0.f, 0.f, 0.f, 0.f, 0.f, 0.f};
    v8f acc[8];
#pragma unroll
    for (int i = 0; i < 8; i++) acc[i] = vzero;

#pragma unroll
    for (int kb = 0; kb < 4; kb++) {
        v16h a = frag_load(&A_sw[(kb * 64 + m0 + lo) * RUA + hi * 16]);
#pragma unroll
        for (int nt = 0; nt < 8; nt++) {
            v16h bf = frag_load(&B_sw[(kb * 128 + nt * 16 + lo) * RUA + hi * 16]);
            acc[nt] = __builtin_amdgcn_wmma_f32_16x16x32_f16(
                false, a, false, bf, (short)0, acc[nt], false, false);
        }
    }
    // C layout: VGPR r -> M = r + 8*hi, N = lane%16.  channels-last store.
#pragma unroll
    for (int nt = 0; nt < 8; nt++)
#pragma unroll
        for (int r = 0; r < 8; r++) {
            int m = m0 + hi * 8 + r;
            out[(size_t)(p0 + m) * DM + nt * 16 + lo] = acc[nt][r];
        }
}

// =====================================================================
// K2: depthwise 3x3 conv + bias + SiLU, channels-last in/out
// =====================================================================
__global__ __launch_bounds__(256) void k_dwconv_silu(const float* __restrict__ xin,
                                                     const float* __restrict__ cw,
                                                     const float* __restrict__ cb,
                                                     float* __restrict__ img)
{
    __shared__ float wsm[DM * 9];
    __shared__ float bsm[DM];
    int tid = threadIdx.x;
    for (int i = tid; i < DM * 9; i += 256) wsm[i] = cw[i];
    for (int i = tid; i < DM; i += 256) bsm[i] = cb[i];
    __syncthreads();

    size_t t = (size_t)blockIdx.x * 256 + tid;
    int c = (int)(t & 127);
    size_t pix = t >> 7;
    int b = (int)(pix >> 14);
    int hw = (int)(pix & (HW - 1));
    int h = hw >> 7, wq = hw & 127;

    float acc = bsm[c];
#pragma unroll
    for (int ky = 0; ky < 3; ky++) {
        int h2 = h + ky - 1;
        if (h2 < 0 || h2 >= HH) continue;
#pragma unroll
        for (int kx = 0; kx < 3; kx++) {
            int w2 = wq + kx - 1;
            if (w2 < 0 || w2 >= WWD) continue;
            acc += xin[((size_t)(b << 14) + (h2 << 7) + w2) * DM + c] * wsm[c * 9 + ky * 3 + kx];
        }
    }
    img[pix * DM + c] = acc / (1.f + expf(-acc));   // SiLU
}

// =====================================================================
// K3: offset branch: depthwise 3x3 -> channel LN -> exact GELU -> 128->16 proj
// =====================================================================
__global__ __launch_bounds__(128) void k_offsets(const float* __restrict__ img,
                                                 const float* __restrict__ dww,
                                                 const float* __restrict__ dwb,
                                                 const float* __restrict__ lng,
                                                 const float* __restrict__ lnb,
                                                 const float* __restrict__ offw,
                                                 const float* __restrict__ offb,
                                                 float* __restrict__ offs)
{
    __shared__ float s1[128], s2[128], gb[128];
    int c = threadIdx.x;
    size_t pix = blockIdx.x;
    int b = (int)(pix >> 14);
    int hw = (int)(pix & (HW - 1));
    int h = hw >> 7, wq = hw & 127;

    float acc = dwb[c];
#pragma unroll
    for (int ky = 0; ky < 3; ky++) {
        int h2 = h + ky - 1;
        if (h2 < 0 || h2 >= HH) continue;
#pragma unroll
        for (int kx = 0; kx < 3; kx++) {
            int w2 = wq + kx - 1;
            if (w2 < 0 || w2 >= WWD) continue;
            acc += img[((size_t)(b << 14) + (h2 << 7) + w2) * DM + c] * dww[c * 9 + ky * 3 + kx];
        }
    }
    s1[c] = acc; s2[c] = acc * acc;
    __syncthreads();
    for (int o = 64; o > 0; o >>= 1) {
        if (c < o) { s1[c] += s1[c + o]; s2[c] += s2[c + o]; }
        __syncthreads();
    }
    float m  = s1[0] * (1.f / 128.f);
    float vv = s2[0] * (1.f / 128.f) - m * m;
    float vn = (acc - m) * rsqrtf(vv + EPSV) * lng[c] + lnb[c];
    float g  = 0.5f * vn * (1.f + erff(vn * 0.70710678118654752f));   // exact GELU
    gb[c] = g;
    __syncthreads();
    if (c < 16) {
        float s = offb[c];
        for (int j = 0; j < 128; j++) s += gb[j] * offw[c * 128 + j];
        offs[pix * 16 + c] = s;
    }
}

// =====================================================================
// K4: DCNv3(k=1) bilinear sample + x_proj (128->10) + dt_w (8->128)
// =====================================================================
__global__ __launch_bounds__(128) void k_dcn_proj(const float* __restrict__ img,
                                                  const float* __restrict__ offs,
                                                  const float* __restrict__ xpw,
                                                  const float* __restrict__ dtw,
                                                  float* __restrict__ xd,
                                                  float* __restrict__ dts,
                                                  float* __restrict__ Bsb,
                                                  float* __restrict__ Csb)
{
    __shared__ float xs_s[128];
    __shared__ float xdbl[10];
    int c = threadIdx.x;
    size_t pix = blockIdx.x;
    int b = (int)(pix >> 14);
    int hw = (int)(pix & (HW - 1));
    int h = hw >> 7, wq = hw & 127;
    int g = c >> 4;

    float ox = offs[pix * 16 + 2 * g];
    float oy = offs[pix * 16 + 2 * g + 1];
    float px = (float)wq + ox, py = (float)h + oy;
    float x0f = floorf(px), y0f = floorf(py);
    float wx = px - x0f, wy = py - y0f;
    int ix0 = (int)x0f, iy0 = (int)y0f;

    auto tap = [&](int yy, int xx) -> float {
        if (yy < 0 || yy >= HH || xx < 0 || xx >= WWD) return 0.f;
        return img[((size_t)(b << 14) + (yy << 7) + xx) * DM + c];
    };
    float v = tap(iy0, ix0)         * (1.f - wy) * (1.f - wx)
            + tap(iy0, ix0 + 1)     * (1.f - wy) * wx
            + tap(iy0 + 1, ix0)     * wy * (1.f - wx)
            + tap(iy0 + 1, ix0 + 1) * wy * wx;

    xd[pix * DM + c] = v;
    xs_s[c] = v;
    __syncthreads();
    if (c < 10) {
        float s = 0.f;
        for (int j = 0; j < 128; j++) s += xpw[c * 128 + j] * xs_s[j];
        xdbl[c] = s;
    }
    __syncthreads();
    float dt = 0.f;
#pragma unroll
    for (int r = 0; r < 8; r++) dt += dtw[c * 8 + r] * xdbl[r];
    dts[pix * DM + c] = dt;
    if (c == 0) { Bsb[pix] = xdbl[8]; Csb[pix] = xdbl[9]; }
}

// =====================================================================
// Selective scan (D_STATE=1): 3-phase chunked scan, lanes = consecutive d
// =====================================================================
__device__ __forceinline__ float softplusf(float x)
{
    return x > 20.f ? x : log1pf(expf(x));
}

__global__ __launch_bounds__(128) void k_scan_p1(const float* __restrict__ dts,
                                                 const float* __restrict__ xs,
                                                 const float* __restrict__ Bsb,
                                                 const float* __restrict__ Alog,
                                                 const float* __restrict__ dtb,
                                                 float* __restrict__ cA,
                                                 float* __restrict__ cH)
{
    int d = threadIdx.x;
    int b = blockIdx.x / NCH, ch = blockIdx.x % NCH;
    float A = -expf(Alog[d]);
    float db = dtb[d];
    float h = 0.f, ap = 1.f;
    int l0 = ch * LC;
    for (int l = l0; l < l0 + LC; l++) {
        size_t pix = (size_t)b * HW + l;
        float delta = softplusf(dts[pix * DM + d] + db);
        float a  = expf(delta * A);
        float bx = delta * Bsb[pix] * xs[pix * DM + d];
        h = a * h + bx;
        ap *= a;
    }
    int o = (b * NCH + ch) * DM + d;
    cA[o] = ap; cH[o] = h;
}

__global__ __launch_bounds__(128) void k_scan_p2(const float* __restrict__ cA,
                                                 const float* __restrict__ cH,
                                                 float* __restrict__ cHin)
{
    int d = threadIdx.x, b = blockIdx.x;
    float hr = 0.f;
    for (int ch = 0; ch < NCH; ch++) {
        int o = (b * NCH + ch) * DM + d;
        cHin[o] = hr;
        hr = cA[o] * hr + cH[o];
    }
}

__global__ __launch_bounds__(128) void k_scan_p3(const float* __restrict__ dts,
                                                 const float* __restrict__ xs,
                                                 const float* __restrict__ Bsb,
                                                 const float* __restrict__ Csb,
                                                 const float* __restrict__ Alog,
                                                 const float* __restrict__ dtb,
                                                 const float* __restrict__ Dsv,
                                                 const float* __restrict__ cHin,
                                                 float* __restrict__ y)
{
    int d = threadIdx.x;
    int b = blockIdx.x / NCH, ch = blockIdx.x % NCH;
    float A = -expf(Alog[d]);
    float db = dtb[d];
    float Dd = Dsv[d];
    float h = cHin[(b * NCH + ch) * DM + d];
    int l0 = ch * LC;
    for (int l = l0; l < l0 + LC; l++) {
        size_t pix = (size_t)b * HW + l;
        float xv = xs[pix * DM + d];
        float delta = softplusf(dts[pix * DM + d] + db);
        float a  = expf(delta * A);
        float bx = delta * Bsb[pix] * xv;
        h = a * h + bx;
        y[pix * DM + d] = h * Csb[pix] + Dd * xv;
    }
}

// =====================================================================
// K7a: channel LayerNorm -> f16 (feeds out_proj WMMA)
// =====================================================================
__global__ __launch_bounds__(128) void k_outln(const float* __restrict__ y,
                                               const float* __restrict__ g,
                                               const float* __restrict__ bb,
                                               _Float16* __restrict__ yh)
{
    __shared__ float s1[128], s2[128];
    int c = threadIdx.x;
    size_t pix = blockIdx.x;
    float v = y[pix * DM + c];
    s1[c] = v; s2[c] = v * v;
    __syncthreads();
    for (int o = 64; o > 0; o >>= 1) {
        if (c < o) { s1[c] += s1[c + o]; s2[c] += s2[c + o]; }
        __syncthreads();
    }
    float m  = s1[0] * (1.f / 128.f);
    float vv = s2[0] * (1.f / 128.f) - m * m;
    yh[pix * DM + c] = (_Float16)((v - m) * rsqrtf(vv + EPSV) * g[c] + bb[c]);
}

// =====================================================================
// K7b: out_proj WMMA GEMM, f16 channels-last in -> f32 NCHW out via LDS transpose
// =====================================================================
#define SMEM_BYTES ((ASW_HALVES + BSW_HALVES) * 2)   // 61440 B; Ct needs 34816 B

__global__ __launch_bounds__(128) void k_outproj(const _Float16* __restrict__ yh,
                                                 const float* __restrict__ w,
                                                 float* __restrict__ out)
{
    __shared__ __align__(16) unsigned char smem[SMEM_BYTES];
    _Float16* A_sw = (_Float16*)smem;
    _Float16* B_sw = (_Float16*)smem + ASW_HALVES;
    float*    Ct   = (float*)smem;   // overlay after compute: 128*68*4 = 34816 B

    const int tid = threadIdx.x;
    const int p0  = blockIdx.x * 64;
    const int b   = p0 >> 14;
    const int hw0 = p0 & (HW - 1);

    for (int idx = tid; idx < 64 * 128; idx += 128) {
        int i = idx >> 7, c = idx & 127;
        int hi, e; a_swizzle(c & 31, hi, e);
        A_sw[((c >> 5) * 64 + i) * RUA + hi * 16 + e] = yh[(size_t)(p0 + i) * DM + c];
    }
    for (int idx = tid; idx < 128 * 128; idx += 128) {
        int n = idx >> 7, k = idx & 127;
        int k32 = k & 31, hi, e; a_swizzle(k32, hi, e);
        B_sw[((k >> 5) * 128 + n) * RUA + hi * 16 + e] = (_Float16)w[idx];
    }
    __syncthreads();

    const int lane = tid & 31, wv = tid >> 5;
    const int m0 = wv * 16;
    const int hi = lane >> 4, lo = lane & 15;

    const v8f vzero = {0.f, 0.f, 0.f, 0.f, 0.f, 0.f, 0.f, 0.f};
    v8f acc[8];
#pragma unroll
    for (int i = 0; i < 8; i++) acc[i] = vzero;

#pragma unroll
    for (int kb = 0; kb < 4; kb++) {
        v16h a = frag_load(&A_sw[(kb * 64 + m0 + lo) * RUA + hi * 16]);
#pragma unroll
        for (int nt = 0; nt < 8; nt++) {
            v16h bf = frag_load(&B_sw[(kb * 128 + nt * 16 + lo) * RUA + hi * 16]);
            acc[nt] = __builtin_amdgcn_wmma_f32_16x16x32_f16(
                false, a, false, bf, (short)0, acc[nt], false, false);
        }
    }
    __syncthreads();   // done reading A_sw/B_sw; overlay Ct
#pragma unroll
    for (int nt = 0; nt < 8; nt++)
#pragma unroll
        for (int r = 0; r < 8; r++) {
            int m = m0 + hi * 8 + r;
            Ct[(nt * 16 + lo) * 68 + m] = acc[nt][r];
        }
    __syncthreads();
    for (int idx = tid; idx < 128 * 64; idx += 128) {
        int n = idx >> 6, i = idx & 63;
        out[((size_t)(b * DM + n)) * HW + hw0 + i] = Ct[n * 68 + i];
    }
}

// =====================================================================
extern "C" void kernel_launch(void* const* d_in, const int* in_sizes, int n_in,
                              void* d_out, int out_size, void* d_ws, size_t ws_size,
                              hipStream_t stream)
{
    (void)in_sizes; (void)n_in; (void)out_size; (void)ws_size;
    const float* x         = (const float*)d_in[0];
    const float* in_proj_w = (const float*)d_in[1];
    const float* conv2d_w  = (const float*)d_in[2];
    const float* conv2d_b  = (const float*)d_in[3];
    const float* dw_w      = (const float*)d_in[4];
    const float* dw_b      = (const float*)d_in[5];
    const float* dw_ln_g   = (const float*)d_in[6];
    const float* dw_ln_b   = (const float*)d_in[7];
    const float* off_w     = (const float*)d_in[8];
    const float* off_b     = (const float*)d_in[9];
    const float* x_proj_w  = (const float*)d_in[10];
    const float* dt_w      = (const float*)d_in[11];
    const float* dt_b      = (const float*)d_in[12];
    const float* A_logs    = (const float*)d_in[13];
    const float* Ds        = (const float*)d_in[14];
    const float* out_ln_g  = (const float*)d_in[15];
    const float* out_ln_b  = (const float*)d_in[16];
    const float* out_proj_w= (const float*)d_in[17];
    float* out = (float*)d_out;

    char* ws = (char*)d_ws;
    const size_t T32 = (size_t)NPIX * DM * sizeof(float);   // 32 MB per tensor
    float* xin  = (float*)(ws + 0);            // K1 out; reused for y after K2
    float* img  = (float*)(ws + T32);          // xc (post-SiLU), needed through K4
    float* xd   = (float*)(ws + 2 * T32);      // xs for scan
    float* dts  = (float*)(ws + 3 * T32);      // dt projection; reused as f16 LN out
    float* offs = (float*)(ws + 4 * T32);      // 4 MB
    char*  tail = ws + 4 * T32 + (size_t)NPIX * 16 * sizeof(float);
    float* Bsb  = (float*)(tail);
    float* Csb  = (float*)(tail + (size_t)NPIX * sizeof(float));
    float* cA   = (float*)(tail + 2 * (size_t)NPIX * sizeof(float));
    float* cH   = (float*)(tail + 2 * (size_t)NPIX * sizeof(float) + 65536);
    float* cHin = (float*)(tail + 2 * (size_t)NPIX * sizeof(float) + 131072);
    float* ybuf = xin;                          // xin dead after K2
    _Float16* yh = (_Float16*)dts;              // dts dead after scan p3

    k_inproj     <<<NPIX / 64, 128, 0, stream>>>(x, in_proj_w, xin);
    k_dwconv_silu<<<(NPIX * DM) / 256, 256, 0, stream>>>(xin, conv2d_w, conv2d_b, img);
    k_offsets    <<<NPIX, 128, 0, stream>>>(img, dw_w, dw_b, dw_ln_g, dw_ln_b,
                                            off_w, off_b, offs);
    k_dcn_proj   <<<NPIX, 128, 0, stream>>>(img, offs, x_proj_w, dt_w,
                                            xd, dts, Bsb, Csb);
    k_scan_p1    <<<N_B * NCH, 128, 0, stream>>>(dts, xd, Bsb, A_logs, dt_b, cA, cH);
    k_scan_p2    <<<N_B, 128, 0, stream>>>(cA, cH, cHin);
    k_scan_p3    <<<N_B * NCH, 128, 0, stream>>>(dts, xd, Bsb, Csb, A_logs, dt_b,
                                                 Ds, cHin, ybuf);
    k_outln      <<<NPIX, 128, 0, stream>>>(ybuf, out_ln_g, out_ln_b, yh);
    k_outproj    <<<NPIX / 64, 128, 0, stream>>>(yh, out_proj_w, out);
}
